// GraphAttention_61890478735565
// MI455X (gfx1250) — matrix operand order repaired and checked by reading, not compiled, not run on hardware
//
#include <hip/hip_runtime.h>
#include <hip/hip_bf16.h>
#include <stdint.h>

typedef __attribute__((ext_vector_type(16))) _Float16 v16h;
typedef __attribute__((ext_vector_type(8)))  _Float16 v8h;
typedef __attribute__((ext_vector_type(4)))  _Float16 v4h;
typedef __attribute__((ext_vector_type(8)))  float    v8f;
typedef __attribute__((ext_vector_type(4)))  unsigned int v4u;
typedef __attribute__((ext_vector_type(8)))  int      v8i;
typedef __attribute__((ext_vector_type(4)))  int      v4i;

#define GB 4
#define GN 2048
#define GF 64
#define GFO 64
#define GH 4
#define NT (GN/16)          // 128 row-tiles per (b,h)
#define ATT_STRIDE 2064     // f16 per LDS att row (2048 + 16 pad), row = 4128 B
#define KCHUNK 256          // k-extent of one TDM-staged feat_neigh^T chunk
#define NCHUNK (GN/KCHUNK)  // 8
#define FNBUF_ELEMS (GFO*KCHUNK)          // 16384 f16 = 32 KiB per buffer
// LDS layout (bytes)
#define OFF_ATT   0
#define OFF_SNL   (16*ATT_STRIDE*2)                   // 66048
#define OFF_MISC  (OFF_SNL + GN*4)                    // 74240
#define OFF_FNB   (OFF_MISC + 192)                    // 74432 (64B aligned)
#define SMEM_BYTES (OFF_FNB + 2*FNBUF_ELEMS*2)        // 139968

// ---------------- TDM: build D# (groups 0/1 per ISA 08_async_tensor §8) and issue
__device__ __forceinline__ void tdm_load_2d(const void* lds_dst, const void* gsrc,
                                            uint32_t data_size_code,   // 0=1B,1=2B,2=4B
                                            uint32_t tensor_d0, uint32_t tensor_d1,
                                            uint32_t tile_d0, uint32_t tile_d1,
                                            uint64_t d0_stride) {
  const uint64_t ga = (uint64_t)(uintptr_t)gsrc;
  const uint32_t la = (uint32_t)(uintptr_t)lds_dst;   // generic LDS addr: low 32 = LDS offset
  v4u g0;
  g0[0] = 1u;                                           // count=1, user descriptor
  g0[1] = la;                                           // lds_addr
  g0[2] = (uint32_t)ga;                                 // global_addr[31:0]
  g0[3] = (uint32_t)((ga >> 32) & 0x01FFFFFFu) | (2u << 30);  // global_addr[56:32] | type=2
  v8i g1;
  g1[0] = (int)(data_size_code << 16);                  // wg_mask=0, data_size, no pad/iterate
  g1[1] = (int)((tensor_d0 & 0xFFFFu) << 16);           // abar_addr=0 | tensor_dim0.lo
  g1[2] = (int)((tensor_d0 >> 16) | ((tensor_d1 & 0xFFFFu) << 16));
  g1[3] = (int)((tensor_d1 >> 16) | (tile_d0 << 16));   // tensor_dim1.hi | tile_dim0
  g1[4] = (int)(tile_d1 & 0xFFFFu);                     // tile_dim1 | tile_dim2=0
  g1[5] = (int)(uint32_t)(d0_stride & 0xFFFFFFFFu);     // tensor_dim0_stride[31:0]
  g1[6] = (int)(uint32_t)((d0_stride >> 32) & 0xFFFFu); // stride[47:32] | dim1_stride=0
  g1[7] = 0;
  const v4i z4 = {0, 0, 0, 0};
#if __clang_major__ >= 23
  const v8i z8 = {0, 0, 0, 0, 0, 0, 0, 0};
  __builtin_amdgcn_tensor_load_to_lds(g0, g1, z4, z4, z8, 0);
#else
  __builtin_amdgcn_tensor_load_to_lds(g0, g1, z4, z4, 0);
#endif
}

// ---------- P0: fold attention vectors into weights: w[h][f] = sum_e W[h,f,e]*a[h,e]
__global__ void gat_p0(const float* __restrict__ Wself, const float* __restrict__ Wneigh,
                       const float* __restrict__ aself, const float* __restrict__ aneigh,
                       float* __restrict__ wsn) {
  const int t = threadIdx.x;          // 256 = H*F
  const int h = t >> 6, f = t & 63;
  const float* ws = Wself  + (h*GF + f)*GFO;
  const float* wn = Wneigh + (h*GF + f)*GFO;
  const float* cs = aself  + h*GFO;
  const float* cn = aneigh + h*GFO;
  float as = 0.f, an = 0.f;
  #pragma unroll 8
  for (int e = 0; e < GFO; ++e) { as += ws[e]*cs[e]; an += wn[e]*cn[e]; }
  wsn[t]       = as;
  wsn[256 + t] = an;
}

// ---------- P1: per-(b,h) 16-row tile: feat_self (f32), feat_neigh^T (f16), scores
__global__ void gat_p1(const float* __restrict__ X,
                       const float* __restrict__ Wself, const float* __restrict__ Wneigh,
                       const float* __restrict__ wsn,
                       float* __restrict__ fs_ws, _Float16* __restrict__ fnT,
                       float* __restrict__ s_self, float* __restrict__ s_neigh) {
  const int lane = threadIdx.x;            // one wave32
  const int row = lane & 15, hi = lane >> 4;
  const int it = blockIdx.x & (NT - 1);
  const int bh = blockIdx.x >> 7;          // NT == 128
  const int h  = bh & (GH - 1);
  const int b  = bh >> 2;
  const int i0 = it * 16;

  // A operand: X[b, i0+row, 0:64] as two 16x32 f16 tiles (ISA 16-bit A layout)
  const float* xrow = X + ((size_t)b*GN + i0 + row)*GF;
  v16h a0, a1;
  #pragma unroll
  for (int e = 0; e < 8; ++e) {
    a0[e]     = (_Float16)xrow[      hi*8 + e];
    a0[e + 8] = (_Float16)xrow[16  + hi*8 + e];
    a1[e]     = (_Float16)xrow[32  + hi*8 + e];
    a1[e + 8] = (_Float16)xrow[48  + hi*8 + e];
  }

  #pragma unroll
  for (int m = 0; m < 2; ++m) {
    const float* Wp = m ? Wneigh : Wself;
    #pragma unroll
    for (int ct = 0; ct < 4; ++ct) {
      const int c0 = ct * 16;
      v16h b0, b1;                         // B operand: 32x16 f16 layout
      #pragma unroll
      for (int e = 0; e < 16; ++e) {
        b0[e] = (_Float16)Wp[(h*GF +      (e>>3)*16 + hi*8 + (e&7))*GFO + c0 + row];
        b1[e] = (_Float16)Wp[(h*GF + 32 + (e>>3)*16 + hi*8 + (e&7))*GFO + c0 + row];
      }
      v8f c = {0.f,0.f,0.f,0.f,0.f,0.f,0.f,0.f};
      c = __builtin_amdgcn_wmma_f32_16x16x32_f16(false, a0, false, b0, (short)0, c, false, false);
      c = __builtin_amdgcn_wmma_f32_16x16x32_f16(false, a1, false, b1, (short)0, c, false, false);
      #pragma unroll
      for (int r = 0; r < 8; ++r) {
        const int M = r + hi*8;            // C/D layout: lanes>=16 hold M+8
        if (m == 0)
          fs_ws[((size_t)bh*GN + i0 + M)*GFO + c0 + row] = c[r];
        else
          fnT[((size_t)bh*GFO + c0 + row)*GN + i0 + M] = (_Float16)c[r];
      }
    }
  }

  // scalar logits: lanes 0-15 -> s_self(node i0+row), lanes 16-31 -> s_neigh
  const float* wv = wsn + hi*256 + h*GFO;
  float acc = 0.f;
  #pragma unroll 8
  for (int f = 0; f < GF; ++f) acc += xrow[f] * wv[f];
  if (hi == 0) s_self[(size_t)bh*GN + i0 + row] = acc;
  else         s_neigh[(size_t)bh*GN + i0 + row] = acc;
}

// ---------- F: fused sigmoid-attention + normalization + TDM-fed aggregation GEMM
__global__ void gat_fused(const float* __restrict__ A,
                          const float* __restrict__ fs_ws, const _Float16* __restrict__ fnT,
                          const float* __restrict__ s_self, const float* __restrict__ s_neigh,
                          float* __restrict__ outp, float* __restrict__ attp) {
  extern __shared__ char smem[];
  _Float16* attl  = (_Float16*)(smem + OFF_ATT);       // 16 x ATT_STRIDE f16
  float* snl      = (float*)(smem + OFF_SNL);          // 2048 f32
  float* rowsum   = (float*)(smem + OFF_MISC);
  float* invv     = rowsum + 16;
  float* ssl      = rowsum + 32;
  _Float16* fnb   = (_Float16*)(smem + OFF_FNB);       // 2 x [64][KCHUNK] f16

  const int t    = threadIdx.x;          // 128 threads = 4 waves
  const int wave = t >> 5;
  const int it = blockIdx.x & (NT - 1);
  const int bh = blockIdx.x >> 7;
  const int h  = bh & (GH - 1);
  const int b  = bh >> 2;
  const int i0 = it * 16;
  const _Float16* fnsrc = fnT + (size_t)bh*GFO*GN;     // [64][2048] f16 for this (b,h)

  // TDM: stage s_neigh (1D, 8 KiB) and feat_neigh^T chunk 0; chunk 0 DMA
  // overlaps the whole sigmoid phase below. TENSORcnt is in-order.
  if (wave == 0) {
    tdm_load_2d(snl, s_neigh + (size_t)bh*GN, /*4B*/2, GN, 1, GN, 1, GN);
    tdm_load_2d(fnb, fnsrc, /*2B*/1, GN, GFO, KCHUNK, GFO, GN);
    __builtin_amdgcn_s_wait_tensorcnt(1);              // s_neigh complete
  }
  if (t < 16) { rowsum[t] = 0.f; ssl[t] = s_self[(size_t)bh*GN + i0 + t]; }
  __syncthreads();

  // Phase 1: stream A once (float4), sigmoid attention into LDS (f16), row sums
  for (int r = 0; r < 16; ++r) {
    const float ss = ssl[r];
    const float* arow = A + ((size_t)b*GN + i0 + r)*GN;
    float acc = 0.f;
    for (int j = t*4; j < GN; j += 512) {
      __builtin_prefetch(arow + j + 2048, 0, 0);       // global_prefetch_b8 (next row)
      const float4 av = *(const float4*)(arow + j);
      const float4 sn = *(const float4*)(snl + j);
      v4h hq;
      #pragma unroll
      for (int q = 0; q < 4; ++q) {
        const float a_q = q == 0 ? av.x : q == 1 ? av.y : q == 2 ? av.z : av.w;
        const float s_q = q == 0 ? sn.x : q == 1 ? sn.y : q == 2 ? sn.z : sn.w;
        float L = ss + s_q;
        L = (L > 0.f) ? L : 0.2f * L;                  // leaky_relu(0.2)
        L += -1e16f * (1.0f - a_q);                    // additive adjacency mask
        const float sg = 1.0f / (1.0f + __expf(-L));
        acc += sg;
        hq[q] = (_Float16)sg;
      }
      *(v4h*)(attl + r*ATT_STRIDE + j) = hq;
    }
    atomicAdd(&rowsum[r], acc);
  }
  __syncthreads();
  if (t < 16) invv[t] = 1.0f / (1.0f + rowsum[t]);
  __syncthreads();

  // Phase 2: normalize, write att (the only global write of the 256 MB tensor)
  float* arow_out = attp + ((size_t)bh*GN + i0)*GN;
  for (int r = 0; r < 16; ++r) {
    const float iv = invv[r];
    for (int j = t*4; j < GN; j += 512) {
      v4h hq = *(v4h*)(attl + r*ATT_STRIDE + j);
      float4 v;
      v.x = (float)hq[0]*iv; v.y = (float)hq[1]*iv;
      v.z = (float)hq[2]*iv; v.w = (float)hq[3]*iv;
      *(float4*)(arow_out + (size_t)r*GN + j) = v;
      hq[0] = (_Float16)v.x; hq[1] = (_Float16)v.y;
      hq[2] = (_Float16)v.z; hq[3] = (_Float16)v.w;
      *(v4h*)(attl + r*ATT_STRIDE + j) = hq;           // scaled copy for the GEMM
    }
  }
  __syncthreads();

  // Phase 3: per-wave 16x16 out tile; K=2048 WMMA, B operand double-buffered via TDM
  const int lane = t & 31;
  const int row = lane & 15, hi = lane >> 4;
  const int c0 = wave * 16;
  const _Float16* arl = attl + row*ATT_STRIDE + hi*16;
  v8f c = {0.f,0.f,0.f,0.f,0.f,0.f,0.f,0.f};
  for (int kc = 0; kc < NCHUNK; ++kc) {
    if (wave == 0) {
      if (kc < NCHUNK - 1) {
        tdm_load_2d(fnb + ((kc + 1) & 1)*FNBUF_ELEMS, fnsrc + (kc + 1)*KCHUNK,
                    /*2B*/1, GN, GFO, KCHUNK, GFO, GN);
        __builtin_amdgcn_s_wait_tensorcnt(1);          // chunk kc done, kc+1 in flight
      } else {
        __builtin_amdgcn_s_wait_tensorcnt(0);
      }
    }
    __syncthreads();                                    // publish chunk kc to all waves
    const _Float16* bbase = fnb + (kc & 1)*FNBUF_ELEMS + (c0 + row)*KCHUNK + hi*16;
    const int kabs = kc*KCHUNK;
    #pragma unroll
    for (int s = 0; s < KCHUNK/32; ++s) {
      const int k0 = s*32;
      const v8h lo = *(const v8h*)(arl + kabs + k0);        // K = .. + hi*16 + 0..7
      const v8h hv = *(const v8h*)(arl + kabs + k0 + 8);    // K = .. + hi*16 + 8..15
      const v16h a = __builtin_shufflevector(lo, hv, 0,1,2,3,4,5,6,7,8,9,10,11,12,13,14,15);
      const v16h bb = *(const v16h*)(bbase + k0);           // 16 consecutive K, N = c0+row
      c = __builtin_amdgcn_wmma_f32_16x16x32_f16(false, a, false, bb, (short)0, c, false, false);
    }
    __syncthreads();                                    // all waves done with buf kc&1
  }
  #pragma unroll
  for (int r = 0; r < 8; ++r) {
    const int M = r + hi*8;
    const float iv = invv[M];
    const float fs = fs_ws[((size_t)bh*GN + i0 + M)*GFO + c0 + row];
    float val = c[r] + fs * iv;                        // feat_self*beta + att@feat_neigh
    val = val > 0.f ? val : 0.f;                       // ReLU on concat output
    outp[((size_t)b*GN + i0 + M)*(GH*GFO) + h*GFO + c0 + row] = val;
  }
}

extern "C" void kernel_launch(void* const* d_in, const int* in_sizes, int n_in,
                              void* d_out, int out_size, void* d_ws, size_t ws_size,
                              hipStream_t stream) {
  const float* X      = (const float*)d_in[0];
  const float* Aa     = (const float*)d_in[1];
  const float* Wself  = (const float*)d_in[2];
  const float* Wneigh = (const float*)d_in[3];
  const float* aself  = (const float*)d_in[4];
  const float* aneigh = (const float*)d_in[5];

  char* w = (char*)d_ws;
  float*    fs_ws   = (float*)w;                                    // 8 MiB  f32 [B,H,N,F_]
  _Float16* fnT     = (_Float16*)(w + (size_t)(8u << 20));          // 4 MiB  f16 [B,H,F_,N]
  float*    s_self  = (float*)(w + (size_t)(12u << 20));            // 128 KiB
  float*    s_neigh = (float*)(w + (size_t)(12u << 20) + (128u << 10));
  float*    wsn     = (float*)(w + (size_t)(12u << 20) + (256u << 10)); // 2 KiB

  float* outp = (float*)d_out;
  float* attp = outp + (size_t)GB * GN * GH * GFO;   // out (2,097,152 f32) then att

  gat_p0   <<<1, 256, 0, stream>>>(Wself, Wneigh, aself, aneigh, wsn);
  gat_p1   <<<GB*GH*NT, 32, 0, stream>>>(X, Wself, Wneigh, wsn, fs_ws, fnT, s_self, s_neigh);
  gat_fused<<<GB*GH*NT, 128, SMEM_BYTES, stream>>>(Aa, fs_ws, fnT, s_self, s_neigh, outp, attp);
}
